// Decoder_13615046328673
// MI455X (gfx1250) — compile-verified, matrix operand order checked by reading, not run on hardware
//
#include <hip/hip_runtime.h>
#include <hip/hip_bf16.h>
#include <math.h>

// ---------------------------------------------------------------------------
// CDNA5 (gfx1250) QA-GNN decoder.
// Dense layers -> v_wmma_f32_16x16x32_bf16 (wave32 WMMA), bias+LN+GELU fused
// in the GEMM epilogue, scatter-mean fused as atomicAdd epilogue.
// Weights (<=96KB) staged into LDS via the Tensor Data Mover, K-loop software-
// pipelined with double-buffered A fragments against ds_load_b128 B reads.
// All control flow around WMMA is uniform (scalar branches only): the ISA
// requires EXEC == all-1s for WMMA.
// ---------------------------------------------------------------------------

typedef __bf16 bf16_t;
typedef __attribute__((ext_vector_type(16))) __bf16 v16bf;
typedef __attribute__((ext_vector_type(8)))  float  v8f;
typedef unsigned int uint32x4 __attribute__((ext_vector_type(4)));
typedef int          int32x8  __attribute__((ext_vector_type(8)));
typedef int          int32x4  __attribute__((ext_vector_type(4)));

struct B256 { uint4 lo, hi; };   // 32 bytes == one 16x(bf16) WMMA fragment

__device__ __forceinline__ bf16_t f2bf(float f) {
  unsigned u = __builtin_bit_cast(unsigned, f);
  unsigned r = u + 0x7FFFu + ((u >> 16) & 1u);     // round-to-nearest-even
  unsigned short h = (unsigned short)(r >> 16);
  return __builtin_bit_cast(bf16_t, h);
}

__device__ __forceinline__ float gelu_exact(float x) {
  return 0.5f * x * (1.0f + erff(x * 0.70710678118654752440f));
}

#define GEMM_LN_GELU 1
#define GEMM_SCATTER 2
#define GEMM_WLDS    4   // stage whole WT in LDS via TDM

// Fully-unrolled per-wave compute: NT 16x16 output tiles, B from LDS.
// No divergent control flow anywhere near the WMMAs.
template<int NT>
__device__ __forceinline__ void wlds_compute(
    const bf16_t* __restrict__ arow, const char* __restrict__ wbuf,
    float* __restrict__ cbuf, const float* __restrict__ bias,
    int N, int K, int wave, int r, int hi, int kb)
{
  const v8f vzero = {0.f, 0.f, 0.f, 0.f, 0.f, 0.f, 0.f, 0.f};
  v8f acc[NT];
#pragma unroll
  for (int j = 0; j < NT; ++j) acc[j] = vzero;

  B256 ta, tn;
  ta.lo = *(const uint4*)(arow + kb);
  ta.hi = *(const uint4*)(arow + kb + 16);
  for (int k0 = 0; k0 < K; k0 += 32) {           // uniform trip count
    tn = ta;
    if (k0 + 32 < K) {                           // uniform (K is a kernel arg)
      tn.lo = *(const uint4*)(arow + k0 + 32 + kb);
      tn.hi = *(const uint4*)(arow + k0 + 32 + kb + 16);
    }
    v16bf af = __builtin_bit_cast(v16bf, ta);
#pragma unroll
    for (int j = 0; j < NT; ++j) {
      int t = wave + 4 * j;                      // data, not control
      const char* brow = wbuf + ((size_t)(t * 16 + r) * (size_t)K + k0 + kb) * 2;
      B256 tb;
      tb.lo = *(const uint4*)brow;               // ds_load_b128
      tb.hi = *(const uint4*)(brow + 32);
      v16bf bfr = __builtin_bit_cast(v16bf, tb);
      acc[j] = __builtin_amdgcn_wmma_f32_16x16x32_bf16(
                   false, af, false, bfr, (short)0, acc[j], false, false);
    }
    ta = tn;
  }
#pragma unroll
  for (int j = 0; j < NT; ++j) {
    int t = wave + 4 * j;
    int ncol = t * 16 + r;
    float bv = bias ? bias[ncol] : 0.f;
#pragma unroll
    for (int rr = 0; rr < 8; ++rr)
      cbuf[(rr + hi * 8) * N + ncol] = acc[j][rr] + bv;
  }
}

// ------------------------------- WMMA GEMM ---------------------------------
// C(M,N) = A(M,K) @ WT(N,K)^T (+bias) [+LN+GELU] [bf16/f32 out | scatter-add]
// Block: 128 threads = 4 waves; one block owns a 16-row M-tile and all of N.
// N/16 must be a multiple of 4 (true for all layers here).
__global__ __launch_bounds__(128)
void wmma_mlp_layer(const bf16_t* __restrict__ A, const bf16_t* __restrict__ WT,
                    const float* __restrict__ bias,
                    const float* __restrict__ gamma, const float* __restrict__ beta,
                    bf16_t* __restrict__ outB, float* __restrict__ outF,
                    int M, int N, int K, int ldO, int flags,
                    const int* __restrict__ sidx, float* __restrict__ sdst, int sld)
{
  extern __shared__ float smem[];
  const bool wlds = (flags & GEMM_WLDS) != 0;
  char*  wbuf = (char*)smem;                                // N*K bf16 (if wlds)
  float* cbuf = smem + (wlds ? ((size_t)N * (size_t)K / 2) : 0);  // 16*N f32
  float* psum = cbuf + 16 * N;      // 128
  float* psq  = psum + 128;         // 128
  float* mrow = psq  + 128;         // 16 means
  float* rrow = mrow + 16;          // 16 rstd

  const int tid  = threadIdx.x;
  const int lane = tid & 31;
  const int wave = tid >> 5;
  const int m0   = blockIdx.x * 16;

  const int r  = lane & 15;
  const int hi = lane >> 4;         // 0: k 0-7,16-23; 1: k 8-15,24-31 (of each 32)
  const int kb = hi * 8;

  const int nTiles = N >> 4;
  const int jTiles = nTiles >> 2;   // tiles per wave (uniform!)
  const bf16_t* arow = A + (size_t)(m0 + r) * (size_t)K;

  if (wlds) {
    // ---- stage WT (N rows x K bf16) into LDS via Tensor Data Mover ----
#if __has_builtin(__builtin_amdgcn_tensor_load_to_lds)
    if (wave == 0) {
      // Tensor DMA descriptor (D#) per CDNA5 ISA ch.8: 2D tile == whole WT.
      unsigned long long ga = (unsigned long long)(uintptr_t)WT;
      unsigned ldsOff = (unsigned)((char*)wbuf - (char*)smem); // dyn-LDS base 0
      uint32x4 g0;
      g0[0] = 1u;                                  // count=1, user descriptor
      g0[1] = ldsOff;                              // lds_addr
      g0[2] = (unsigned)(ga & 0xFFFFFFFFu);        // global_addr[31:0]
      g0[3] = (unsigned)((ga >> 32) & 0x01FFFFFFu) | 0x80000000u; // [56:32]|type=2
      int32x8 g1;
      g1[0] = 0x10000;                             // data_size=1 (2 bytes)
      g1[1] = (int)((K & 0xFFFF) << 16);           // tensor_dim0[15:0] @ bits63:48
      g1[2] = (int)(((unsigned)K >> 16) | ((unsigned)(N & 0xFFFF) << 16));
      g1[3] = (int)(((unsigned)N >> 16) | ((unsigned)(K & 0xFFFF) << 16)); // tile_dim0=K
      g1[4] = (int)(N & 0xFFFF);                   // tile_dim1=N, tile_dim2=0
      g1[5] = (int)K;                              // tensor_dim0_stride
      g1[6] = 0;
      g1[7] = 0;
      int32x4 z4 = {0, 0, 0, 0};
      int32x8 z8 = {0, 0, 0, 0, 0, 0, 0, 0};
      __builtin_amdgcn_tensor_load_to_lds(g0, g1, z4, z4, z8, 0);
      __builtin_amdgcn_s_wait_tensorcnt(0);
    }
#else
    for (int i = tid; i < (N * K) / 8; i += 128)
      ((uint4*)wbuf)[i] = ((const uint4*)WT)[i];
#endif
    __syncthreads();

    // ---- uniform dispatch on tiles-per-wave; WMMA with EXEC all-1s ----
    if (jTiles == 2)      wlds_compute<2>(arow, wbuf, cbuf, bias, N, K, wave, r, hi, kb);
    else if (jTiles == 4) wlds_compute<4>(arow, wbuf, cbuf, bias, N, K, wave, r, hi, kb);
    else if (jTiles == 8) wlds_compute<8>(arow, wbuf, cbuf, bias, N, K, wave, r, hi, kb);
    else if (jTiles == 1) wlds_compute<1>(arow, wbuf, cbuf, bias, N, K, wave, r, hi, kb);
  } else {
    // ---- large-K path: B streamed from global; uniform j loop ----
    for (int j = 0; j < jTiles; ++j) {
      const int t = wave + 4 * j;
      const bf16_t* brow = WT + (size_t)(t * 16 + r) * (size_t)K;
      v8f acc = {0.f, 0.f, 0.f, 0.f, 0.f, 0.f, 0.f, 0.f};
      for (int k0 = 0; k0 < K; k0 += 32) {
        if (k0 + 64 < K) {
          __builtin_prefetch(arow + k0 + 64, 0, 1);   // global_prefetch_b8
          __builtin_prefetch(brow + k0 + 64, 0, 1);
        }
        B256 ta, tb;
        ta.lo = *(const uint4*)(arow + k0 + kb);
        ta.hi = *(const uint4*)(arow + k0 + kb + 16);
        tb.lo = *(const uint4*)(brow + k0 + kb);
        tb.hi = *(const uint4*)(brow + k0 + kb + 16);
        v16bf af  = __builtin_bit_cast(v16bf, ta);
        v16bf bfr = __builtin_bit_cast(v16bf, tb);
        acc = __builtin_amdgcn_wmma_f32_16x16x32_bf16(
                  false, af, false, bfr, (short)0, acc, false, false);
      }
      const int ncol = t * 16 + r;
      const float bv = bias ? bias[ncol] : 0.f;
#pragma unroll
      for (int rr = 0; rr < 8; ++rr)
        cbuf[(rr + hi * 8) * N + ncol] = acc[rr] + bv;
    }
  }
  __syncthreads();

  const bool ln = (flags & GEMM_LN_GELU) != 0;
  if (ln) {
    const int rr = tid >> 3;        // 0..15 : row
    const int ss = tid & 7;         // 0..7  : slice
    float s = 0.f, q = 0.f;
    for (int n = ss; n < N; n += 8) {
      float v = cbuf[rr * N + n];
      s += v; q += v * v;
    }
    psum[tid] = s; psq[tid] = q;
    __syncthreads();
    if (tid < 16) {
      float s2 = 0.f, q2 = 0.f;
      for (int j = 0; j < 8; ++j) { s2 += psum[tid * 8 + j]; q2 += psq[tid * 8 + j]; }
      float mean = s2 / (float)N;
      float var  = q2 / (float)N - mean * mean;
      mrow[tid] = mean;
      rrow[tid] = rsqrtf(var + 1e-5f);
    }
    __syncthreads();
  }

  for (int i = tid; i < 16 * N; i += 128) {
    int rr = i / N, n = i - rr * N;
    float v = cbuf[i];
    if (ln) {
      v = (v - mrow[rr]) * rrow[rr];
      v = v * gamma[n] + beta[n];
      v = gelu_exact(v);
    }
    size_t go = (size_t)(m0 + rr) * (size_t)ldO + n;
    if (outB) outB[go] = f2bf(v);
    if (outF) outF[go] = v;
    if (flags & GEMM_SCATTER) {
      int d = sidx[m0 + rr];
      atomicAdd(&sdst[(size_t)d * (size_t)sld + n], v);
    }
  }
}

// ------------------------------ small kernels ------------------------------

__global__ void zero_f32(float* p, size_t n) {
  size_t i = (size_t)blockIdx.x * blockDim.x + threadIdx.x;
  if (i < n) p[i] = 0.f;
}
__global__ void zero_i32(int* p, size_t n) {
  size_t i = (size_t)blockIdx.x * blockDim.x + threadIdx.x;
  if (i < n) p[i] = 0;
}
__global__ void convert_bf16(const float* __restrict__ s, bf16_t* __restrict__ d, size_t n) {
  size_t i = (size_t)blockIdx.x * blockDim.x + threadIdx.x;
  if (i < n) d[i] = f2bf(s[i]);
}
// W (K,N) f32 row-major  ->  WT (N,K) bf16 row-major
__global__ void transpose_bf16(const float* __restrict__ W, bf16_t* __restrict__ WT,
                               int K, int N) {
  size_t i = (size_t)blockIdx.x * blockDim.x + threadIdx.x;
  if (i >= (size_t)K * N) return;
  int k = (int)(i / N), n = (int)(i - (size_t)k * N);
  WT[(size_t)n * K + k] = f2bf(W[i]);
}
__global__ void scan_offsets(const int* __restrict__ nt, int* __restrict__ offs, int n) {
  if (blockIdx.x == 0 && threadIdx.x == 0) {
    int acc = 0;
    for (int i = 0; i < n; ++i) { offs[i] = acc; acc += nt[i]; }
  }
}
__global__ void count_col(const int* __restrict__ col, int* __restrict__ cnt, int E) {
  int e = blockIdx.x * blockDim.x + threadIdx.x;
  if (e < E) atomicAdd(&cnt[col[e]], 1);
}
// Xe(E,384) = [x[row] | x[col] | rel];  IN2(E,0:128) = x[row]
__global__ void gather_edges(const int* __restrict__ row, const int* __restrict__ col,
                             const int* __restrict__ c_ids, const int* __restrict__ rel_ids,
                             const float* __restrict__ ct, const float* __restrict__ rt,
                             bf16_t* __restrict__ Xe, bf16_t* __restrict__ IN2, int E)
{
  size_t i = (size_t)blockIdx.x * blockDim.x + threadIdx.x;
  if (i >= (size_t)E * 384) return;
  int e = (int)(i / 384), c = (int)(i - (size_t)e * 384);
  float v;
  if (c < 128) {
    v = ct[(size_t)c_ids[row[e]] * 128 + c];
    IN2[(size_t)e * 256 + c] = f2bf(v);
  } else if (c < 256) {
    v = ct[(size_t)c_ids[col[e]] * 128 + (c - 128)];
  } else {
    v = rt[(size_t)rel_ids[e] * 128 + (c - 256)];
  }
  Xe[i] = f2bf(v);
}
// X2(N,384) = [concept | agg/max(cnt,1)]
__global__ void build_node_in(const int* __restrict__ c_ids, const float* __restrict__ ct,
                              const float* __restrict__ agg, const int* __restrict__ cnt,
                              bf16_t* __restrict__ X2, int Nn)
{
  size_t i = (size_t)blockIdx.x * blockDim.x + threadIdx.x;
  if (i >= (size_t)Nn * 384) return;
  int n = (int)(i / 384), c = (int)(i - (size_t)n * 384);
  float v = (c < 128) ? ct[(size_t)c_ids[n] * 128 + c]
                      : agg[(size_t)n * 256 + (c - 128)] / fmaxf((float)cnt[n], 1.f);
  X2[i] = f2bf(v);
}
// QA(BS,L,128) = gelu(ragged node_vecs padded to L) as bf16
__global__ void build_qa(const float* __restrict__ NV, const int* __restrict__ offs,
                         const int* __restrict__ ntup, bf16_t* __restrict__ QA,
                         int Nn, int L)
{
  size_t i = (size_t)blockIdx.x * blockDim.x + threadIdx.x;
  size_t total = (size_t)512 * L * 128;
  if (i >= total) return;
  int d = (int)(i & 127);
  size_t bt = i >> 7;
  int t = (int)(bt % L), b = (int)(bt / L);
  int ntb = ntup[b];
  int lim = ntb < L ? ntb : L;
  float v = 0.f;
  if (t < lim) {
    long idx = (long)offs[b] + t;
    if (idx < 0) idx = 0;
    if (idx > Nn - 1) idx = Nn - 1;
    v = gelu_exact(NV[(size_t)idx * 128 + d]);
  }
  QA[i] = f2bf(v);
}
// Masked 2-head softmax pooling; att scores to d_out tail, OUTIN=[pooled|sent].
__global__ __launch_bounds__(256)
void attn_pool(const float* __restrict__ Q, const float* __restrict__ Kb,
               const float* __restrict__ Vb, const int* __restrict__ ntup,
               const float* __restrict__ sent, float* __restrict__ out,
               bf16_t* __restrict__ OUTIN, int BSc, int L)
{
  __shared__ float qs[128];
  __shared__ float sc[2 * 200];
  __shared__ float rsum[2];
  int b = blockIdx.x, tid = threadIdx.x;
  if (tid < 128) qs[tid] = Q[(size_t)b * 128 + tid];
  __syncthreads();
  int ntb = ntup[b];
  for (int i = tid; i < 2 * L; i += blockDim.x) {
    int h = i / L, t = i - h * L;
    bool valid = (t < ntb) || (ntb <= 0 && t == 0);
    float s = -INFINITY;
    if (valid) {
      const float* krow = Kb + ((size_t)b * L + t) * 128 + h * 64;
      float d = 0.f;
      for (int j = 0; j < 64; ++j) d += qs[h * 64 + j] * krow[j];
      s = d * 0.125f;                       // / sqrt(64)
    }
    sc[i] = s;
  }
  __syncthreads();
  if (tid < 2) {
    int h = tid;
    float mx = -INFINITY;
    for (int t = 0; t < L; ++t) mx = fmaxf(mx, sc[h * L + t]);
    float sum = 0.f;
    for (int t = 0; t < L; ++t) { float e = __expf(sc[h * L + t] - mx); sc[h * L + t] = e; sum += e; }
    rsum[h] = 1.f / sum;
  }
  __syncthreads();
  for (int i = tid; i < 2 * L; i += blockDim.x) {
    int h = i / L, t = i - h * L;
    float a = sc[i] * rsum[h];
    sc[i] = a;
    out[BSc + ((size_t)h * BSc + b) * L + t] = a;   // att_scores after logits
  }
  __syncthreads();
  for (int d = tid; d < 128; d += blockDim.x) {
    int h = d >> 6;
    float p = 0.f;
    for (int t = 0; t < L; ++t)
      p += sc[h * L + t] * Vb[((size_t)b * L + t) * 128 + d];
    OUTIN[(size_t)b * 1152 + d] = f2bf(p);
  }
  for (int j = tid; j < 1024; j += blockDim.x)
    OUTIN[(size_t)b * 1152 + 128 + j] = f2bf(sent[(size_t)b * 1024 + j]);
}
// final (512,512) @ (512,1) + b -> logits
__global__ __launch_bounds__(128)
void final_dot(const bf16_t* __restrict__ OH, const float* __restrict__ W1,
               const float* __restrict__ b1, float* __restrict__ out, int Kd)
{
  __shared__ float red[128];
  int m = blockIdx.x, tid = threadIdx.x;
  float s = 0.f;
  for (int k = tid; k < Kd; k += 128) s += (float)OH[(size_t)m * Kd + k] * W1[k];
  red[tid] = s; __syncthreads();
  for (int off = 64; off > 0; off >>= 1) {
    if (tid < off) red[tid] += red[tid + off];
    __syncthreads();
  }
  if (tid == 0) out[m] = red[0] + b1[0];
}

// ------------------------------- host driver -------------------------------
// Input order = setup_inputs() dict order, nested params flattened as pytree
// leaves (dict keys alphabetical).
enum {
  IN_SENT = 0, IN_EDGE_INDEX, IN_CIDS, IN_U, IN_BATCH, IN_NTUP, IN_RELIDS,
  IN_MAXT, IN_CTAB, IN_RTAB,
  E_W0 = 10, E_W1, E_W2, E_B0, E_B1, E_B2, E_BE0, E_BE1, E_G0, E_G1,
  N1_W0 = 20, N1_W1, N1_W2, N1_B0, N1_B1, N1_B2, N1_BE0, N1_BE1, N1_G0, N1_G1,
  N2_W0 = 30, N2_W1, N2_W2, N2_B0, N2_B1, N2_B2, N2_BE0, N2_BE1, N2_G0, N2_G1,
  A_WK = 40, A_WQ, A_WV, A_BK, A_BQ, A_BV,
  O_W0 = 46, O_W1, O_B0, O_B1, O_BE0, O_G0
};

static inline unsigned nblk(size_t n, int b) { return (unsigned)((n + b - 1) / b); }

extern "C" void kernel_launch(void* const* d_in, const int* in_sizes, int n_in,
                              void* d_out, int out_size, void* d_ws, size_t ws_size,
                              hipStream_t stream)
{
  const int E  = in_sizes[IN_RELIDS];     // 307200
  const int Nn = in_sizes[IN_CIDS];       // 76800
  const int BS = in_sizes[IN_NTUP];       // 512
  const int L  = 200;                      // max_tuple_num

  const float* sent = (const float*)d_in[IN_SENT];
  const int* eidx   = (const int*)d_in[IN_EDGE_INDEX];
  const int* row    = eidx;
  const int* col    = eidx + E;
  const int* cids   = (const int*)d_in[IN_CIDS];
  const int* ntup   = (const int*)d_in[IN_NTUP];
  const int* relids = (const int*)d_in[IN_RELIDS];
  const float* ctab = (const float*)d_in[IN_CTAB];
  const float* rtab = (const float*)d_in[IN_RTAB];
  float* out = (float*)d_out;

  // ---------------- workspace carve-up ----------------
  char* p = (char*)d_ws;
  auto alloc = [&](size_t bytes) -> char* {
    char* r = p; p += (bytes + 255) & ~(size_t)255; return r;
  };
  bf16_t* Xe   = (bf16_t*)alloc((size_t)E * 384 * 2);
  bf16_t* IN2  = (bf16_t*)alloc((size_t)E * 256 * 2);
  bf16_t* H1   = (bf16_t*)alloc((size_t)E * 128 * 2);
  bf16_t* H2   = (bf16_t*)alloc((size_t)E * 128 * 2);
  float*  agg  = (float*)alloc((size_t)Nn * 256 * 4);
  int*    cnt  = (int*)alloc((size_t)Nn * 4);
  bf16_t* X2   = (bf16_t*)alloc((size_t)Nn * 384 * 2);
  bf16_t* NH1  = (bf16_t*)alloc((size_t)Nn * 128 * 2);
  bf16_t* NH2  = (bf16_t*)alloc((size_t)Nn * 128 * 2);
  float*  NV   = (float*)alloc((size_t)Nn * 128 * 4);
  bf16_t* QA   = (bf16_t*)alloc((size_t)BS * L * 128 * 2);
  float*  Kbuf = (float*)alloc((size_t)BS * L * 128 * 4);
  float*  Vbuf = (float*)alloc((size_t)BS * L * 128 * 4);
  float*  Qbuf = (float*)alloc((size_t)BS * 128 * 4);
  bf16_t* sentB = (bf16_t*)alloc((size_t)BS * 1024 * 2);
  bf16_t* OUTIN = (bf16_t*)alloc((size_t)BS * 1152 * 2);
  bf16_t* OH    = (bf16_t*)alloc((size_t)BS * 512 * 2);
  int*    offs  = (int*)alloc((size_t)BS * 4);
  bf16_t* WTe0 = (bf16_t*)alloc((size_t)128 * 384 * 2);
  bf16_t* WTe1 = (bf16_t*)alloc((size_t)128 * 128 * 2);
  bf16_t* WTe2 = (bf16_t*)alloc((size_t)128 * 128 * 2);
  bf16_t* WTn10 = (bf16_t*)alloc((size_t)128 * 256 * 2);
  bf16_t* WTn11 = (bf16_t*)alloc((size_t)128 * 128 * 2);
  bf16_t* WTn12 = (bf16_t*)alloc((size_t)256 * 128 * 2);
  bf16_t* WTn20 = (bf16_t*)alloc((size_t)128 * 384 * 2);
  bf16_t* WTn21 = (bf16_t*)alloc((size_t)128 * 128 * 2);
  bf16_t* WTn22 = (bf16_t*)alloc((size_t)128 * 128 * 2);
  bf16_t* WTq  = (bf16_t*)alloc((size_t)128 * 1024 * 2);
  bf16_t* WTk  = (bf16_t*)alloc((size_t)128 * 128 * 2);
  bf16_t* WTv  = (bf16_t*)alloc((size_t)128 * 128 * 2);
  bf16_t* WTo0 = (bf16_t*)alloc((size_t)512 * 1152 * 2);
  (void)ws_size; (void)n_in; (void)out_size;

  auto F = [&](int i) { return (const float*)d_in[i]; };

  // ---------------- weight prep ----------------
  auto tr = [&](int widx, int K, int N, bf16_t* dst) {
    transpose_bf16<<<nblk((size_t)K * N, 256), 256, 0, stream>>>(F(widx), dst, K, N);
  };
  tr(E_W0, 384, 128, WTe0); tr(E_W1, 128, 128, WTe1); tr(E_W2, 128, 128, WTe2);
  tr(N1_W0, 256, 128, WTn10); tr(N1_W1, 128, 128, WTn11); tr(N1_W2, 128, 256, WTn12);
  tr(N2_W0, 384, 128, WTn20); tr(N2_W1, 128, 128, WTn21); tr(N2_W2, 128, 128, WTn22);
  tr(A_WQ, 1024, 128, WTq); tr(A_WK, 128, 128, WTk); tr(A_WV, 128, 128, WTv);
  tr(O_W0, 1152, 512, WTo0);
  convert_bf16<<<nblk((size_t)BS * 1024, 256), 256, 0, stream>>>(sent, sentB, (size_t)BS * 1024);

  // ---------------- graph prep ----------------
  zero_f32<<<nblk((size_t)Nn * 256, 256), 256, 0, stream>>>(agg, (size_t)Nn * 256);
  zero_i32<<<nblk((size_t)Nn, 256), 256, 0, stream>>>(cnt, (size_t)Nn);
  scan_offsets<<<1, 32, 0, stream>>>(ntup, offs, BS);
  gather_edges<<<nblk((size_t)E * 384, 256), 256, 0, stream>>>(
      row, col, cids, relids, ctab, rtab, Xe, IN2, E);
  count_col<<<nblk((size_t)E, 256), 256, 0, stream>>>(col, cnt, E);

  auto gemm = [&](const bf16_t* A, const bf16_t* WT, const float* bias,
                  const float* g, const float* be, bf16_t* oB, float* oF,
                  int M, int N, int K, int ldO, int flags,
                  const int* sidx, float* sdst, int sld) {
    size_t wbytes = (size_t)N * K * 2;
    if (wbytes <= 96 * 1024) flags |= GEMM_WLDS; else wbytes = 0;
    size_t smem = wbytes + (size_t)(16 * N + 288) * sizeof(float);
    wmma_mlp_layer<<<M / 16, 128, smem, stream>>>(
        A, WT, bias, g, be, oB, oF, M, N, K, ldO, flags, sidx, sdst, sld);
  };

  // ---------------- Edge MLP: (E,384)->128->128->128 ----------------
  gemm(Xe, WTe0, F(E_B0), F(E_G0), F(E_BE0), H1, nullptr, E, 128, 384, 128,
       GEMM_LN_GELU, nullptr, nullptr, 0);
  gemm(H1, WTe1, F(E_B1), F(E_G1), F(E_BE1), H2, nullptr, E, 128, 128, 128,
       GEMM_LN_GELU, nullptr, nullptr, 0);
  gemm(H2, WTe2, F(E_B2), nullptr, nullptr, IN2 + 128, nullptr, E, 128, 128, 256,
       0, nullptr, nullptr, 0);                 // eattr' -> IN2[:,128:256]

  // ---------------- Node message MLP: (E,256)->128->128->256, scatter-add --
  gemm(IN2, WTn10, F(N1_B0), F(N1_G0), F(N1_BE0), H1, nullptr, E, 128, 256, 128,
       GEMM_LN_GELU, nullptr, nullptr, 0);
  gemm(H1, WTn11, F(N1_B1), F(N1_G1), F(N1_BE1), H2, nullptr, E, 128, 128, 128,
       GEMM_LN_GELU, nullptr, nullptr, 0);
  gemm(H2, WTn12, F(N1_B2), nullptr, nullptr, nullptr, nullptr, E, 256, 128, 256,
       GEMM_SCATTER, col, agg, 256);            // msg -> segment_sum(agg)

  // ---------------- Node MLP: (N,384)->128->128->128 ----------------
  build_node_in<<<nblk((size_t)Nn * 384, 256), 256, 0, stream>>>(cids, ctab, agg, cnt, X2, Nn);
  gemm(X2, WTn20, F(N2_B0), F(N2_G0), F(N2_BE0), NH1, nullptr, Nn, 128, 384, 128,
       GEMM_LN_GELU, nullptr, nullptr, 0);
  gemm(NH1, WTn21, F(N2_B1), F(N2_G1), F(N2_BE1), NH2, nullptr, Nn, 128, 128, 128,
       GEMM_LN_GELU, nullptr, nullptr, 0);
  gemm(NH2, WTn22, F(N2_B2), nullptr, nullptr, nullptr, NV, Nn, 128, 128, 128,
       0, nullptr, nullptr, 0);

  // ---------------- Ragged pad + GELU, attention pooling ----------------
  build_qa<<<nblk((size_t)BS * L * 128, 256), 256, 0, stream>>>(NV, offs, ntup, QA, Nn, L);
  gemm(sentB, WTq, F(A_BQ), nullptr, nullptr, nullptr, Qbuf, BS, 128, 1024, 128,
       0, nullptr, nullptr, 0);
  gemm(QA, WTk, F(A_BK), nullptr, nullptr, nullptr, Kbuf, BS * L, 128, 128, 128,
       0, nullptr, nullptr, 0);
  gemm(QA, WTv, F(A_BV), nullptr, nullptr, nullptr, Vbuf, BS * L, 128, 128, 128,
       0, nullptr, nullptr, 0);
  attn_pool<<<BS, 256, 0, stream>>>(Qbuf, Kbuf, Vbuf, ntup, sent, out, OUTIN, BS, L);

  // ---------------- Output MLP: (BS,1152)->512 (LN+GELU) -> 1 ----------------
  gemm(OUTIN, WTo0, F(O_B0), F(O_G0), F(O_BE0), OH, nullptr, BS, 512, 1152, 512,
       GEMM_LN_GELU, nullptr, nullptr, 0);
  final_dot<<<BS, 128, 0, stream>>>(OH, F(O_W1), F(O_B1), out, 512);
}
